// DkNetCL_4784593568093
// MI455X (gfx1250) — matrix operand order repaired and checked by reading, not compiled, not run on hardware
//
#include <hip/hip_runtime.h>
#include <hip/hip_bf16.h>

// ---------------------------------------------------------------------------
// DkNetCL on gfx1250: 4x [3x3 conv + SRePro] + FC(32768->10), B=1024, 32x32.
//
// SRePro scales commute through the linear convs, so all convs run on the
// unscaled tensors; per-sample sum-of-squares S_l is fused into the conv
// epilogue and the cumulative scalar c[b] is applied once in the FC kernel.
//
// Conv = implicit GEMM on v_wmma_f32_16x16x32_bf16 (wave32):
//   per wave: 16 pixels x 32 couts, K = 9 taps x 32 channels (layer 0 padded).
// Activations: zero-haloed NHWC [B][34][34][32] bf16 (branch-free tap loads,
// all 9 tap displacements folded into global_load_b128 immediate offsets).
// Weights: pre-packed into exact B-fragment lane layout, staged in LDS.
// ---------------------------------------------------------------------------

typedef __attribute__((ext_vector_type(16))) __bf16 v16bf;
typedef __attribute__((ext_vector_type(8)))  float  v8f;

#define HP 34                      // padded spatial dim
#define ACT_ELEMS (1024u * HP * HP * 32u)          // 37,879,808 bf16 elems
#define ACT_BYTES ((size_t)ACT_ELEMS * 2u)         // 75,759,616 B
#define WPACK_PER_LAYER (9 * 2 * 32 * 16)          // 9216 shorts / layer

struct Pack32 { uint4 lo, hi; };

static __device__ __forceinline__ v16bf make_frag(uint4 lo, uint4 hi) {
  Pack32 t{lo, hi};
  return __builtin_bit_cast(v16bf, t);
}

static __device__ __forceinline__ unsigned short f2bf(float f) {
  unsigned int u = __builtin_bit_cast(unsigned int, f);
  u += 0x7FFFu + ((u >> 16) & 1u);               // round-to-nearest-even
  return (unsigned short)(u >> 16);
}

static __device__ __forceinline__ float bf2f(unsigned short s) {
  unsigned int u = ((unsigned int)s) << 16;
  return __builtin_bit_cast(float, u);
}

// --------------------------- zero workspace --------------------------------
__global__ __launch_bounds__(256) void zero_kernel(uint4* p, unsigned int n16) {
  unsigned int i = blockIdx.x * blockDim.x + threadIdx.x;
  unsigned int stride = gridDim.x * blockDim.x;
  uint4 z{0u, 0u, 0u, 0u};
  for (; i < n16; i += stride) p[i] = z;
}

// ------------------- pack input NCHW f32 -> padded NHWC bf16 ----------------
__global__ __launch_bounds__(256) void pack_input_kernel(
    const float* __restrict__ x, unsigned short* __restrict__ act) {
  unsigned int idx = blockIdx.x * blockDim.x + threadIdx.x;   // B*32*32 = 1M
  if (idx >= 1024u * 1024u) return;
  int bb = idx >> 10, rem = idx & 1023, y = rem >> 5, xx = rem & 31;
  const float* src = x + (size_t)bb * 3 * 1024 + y * 32 + xx;
  unsigned short* dst = act + ((size_t)((bb * HP + y + 1) * HP + xx + 1)) * 32;
  dst[0] = f2bf(src[0]);
  dst[1] = f2bf(src[1024]);
  dst[2] = f2bf(src[2048]);
  // channels 3..31 stay zero from the memset pass
}

// ---------- pack conv weights into B-fragment lane layout (bf16) -----------
// wpack[l][tap][tile][lane][e]; K=(e&7)+16*(e>>3)+8*(lane>=16); N=(lane&15)+16*tile
__global__ __launch_bounds__(256) void pack_weights_kernel(
    const float* __restrict__ w0, const float* __restrict__ w1,
    const float* __restrict__ w2, const float* __restrict__ w3,
    unsigned short* __restrict__ wpack) {
  unsigned int idx = blockIdx.x * blockDim.x + threadIdx.x;   // 36864 total
  if (idx >= 4u * WPACK_PER_LAYER) return;
  int l    = idx / WPACK_PER_LAYER;
  int r    = idx % WPACK_PER_LAYER;
  int tap  = r / 1024;
  int r2   = r % 1024;
  int tile = r2 >> 9;
  int r3   = r2 & 511;
  int lane = r3 >> 4;
  int e    = r3 & 15;
  int k    = (e & 7) + ((e >> 3) << 4) + ((lane >= 16) ? 8 : 0);        // input channel
  int n    = (lane & 15) + 16 * tile;            // output channel
  int dy = tap / 3, dx = tap % 3;
  const float* w = (l == 0) ? w0 : (l == 1) ? w1 : (l == 2) ? w2 : w3;
  int cin = (l == 0) ? 3 : 32;
  float v = (k < cin) ? w[((n * cin + k) * 3 + dy) * 3 + dx] : 0.0f;
  wpack[idx] = f2bf(v);
}

// ------------------------------- conv layer --------------------------------
__global__ __launch_bounds__(256) void conv_kernel(
    const unsigned short* __restrict__ in, unsigned short* __restrict__ out,
    const unsigned short* __restrict__ wpack, float* __restrict__ S) {
  __shared__ __align__(32) unsigned short wlds[WPACK_PER_LAYER];
  {
    const unsigned int* src = (const unsigned int*)wpack;
    unsigned int* dst = (unsigned int*)wlds;
#pragma unroll
    for (int i = 0; i < 18; ++i)                 // 18*256*4B = 18432 B
      dst[threadIdx.x + 256 * i] = src[threadIdx.x + 256 * i];
  }
  __syncthreads();

  int wave = threadIdx.x >> 5, lane = threadIdx.x & 31;
  int gw = blockIdx.x * 8 + wave;                // 65536 waves total
  int bb = gw >> 6;                              // sample
  int rem = gw & 63;
  int y   = rem >> 1;                            // output row 0..31
  int x0  = (rem & 1) << 4;                      // 0 or 16
  int m   = lane & 15;                           // A-matrix row (pixel)
  int ch0 = (lane >> 4) << 3;                    // 0 or 8 (A K-halves)

  // One per-lane base pointer; all 9 tap displacements are compile-time
  // constants (dy*HP+dx)*32 elems -> folded into global_load IOFFSET.
  const unsigned short* abase =
      in + ((size_t)((bb * HP + y) * HP + x0 + m)) * 32 + ch0;
  const unsigned short* bbase = wlds + lane * 16;  // + (tap*2+t)*512 immediate

  v8f acc0 = {}, acc1 = {};
#pragma unroll
  for (int dy = 0; dy < 3; ++dy) {
#pragma unroll
    for (int dx = 0; dx < 3; ++dx) {
      const unsigned short* ap = abase + (dy * HP + dx) * 32;
      uint4 lo = *(const uint4*)ap;              // channels {0-7 | 8-15}
      uint4 hi = *(const uint4*)(ap + 16);       // channels {16-23 | 24-31}
      v16bf a = make_frag(lo, hi);
      int tap = dy * 3 + dx;
      v16bf bf0 = *(const v16bf*)(bbase + (tap * 2 + 0) * 512);
      v16bf bf1 = *(const v16bf*)(bbase + (tap * 2 + 1) * 512);
      acc0 = __builtin_amdgcn_wmma_f32_16x16x32_bf16(
          false, a, false, bf0, (short)0, acc0, false, false);
      acc1 = __builtin_amdgcn_wmma_f32_16x16x32_bf16(
          false, a, false, bf1, (short)0, acc1, false, false);
    }
  }

  // fused per-sample sum-of-squares of the raw conv output
  float s = 0.0f;
#pragma unroll
  for (int r = 0; r < 8; ++r) s += acc0[r] * acc0[r] + acc1[r] * acc1[r];
#pragma unroll
  for (int msk = 16; msk; msk >>= 1) s += __shfl_xor(s, msk, 32);
  if (lane == 0) atomicAdd(&S[bb], s);

  // store D: lanes 0-15 -> M=r, lanes 16-31 -> M=r+8; N = lane&15 (+16 tile)
  int mrow = (lane >> 4) << 3;
  int n = lane & 15;
  unsigned short* obase =
      out + ((size_t)((bb * HP + y + 1) * HP + x0 + mrow + 1)) * 32 + n;
#pragma unroll
  for (int r = 0; r < 8; ++r) {
    obase[r * 32]      = f2bf(acc0[r]);          // +r pixels, cout n
    obase[r * 32 + 16] = f2bf(acc1[r]);          // +r pixels, cout n+16
  }
}

// -------------------- cumulative SRePro scale per sample -------------------
__global__ __launch_bounds__(256) void scale_kernel(
    const float* __restrict__ S, float* __restrict__ cbuf) {
  unsigned int bb = blockIdx.x * blockDim.x + threadIdx.x;
  if (bb >= 1024) return;
  float c = 1.0f;
#pragma unroll
  for (int l = 0; l < 4; ++l) {
    float sq = c * c * S[l * 1024 + bb];         // ||c_{l-1} * y_l||^2
    c = 2.0f * c / (2.0f + sq);
  }
  cbuf[bb] = c;
}

// --------------------------------- FC --------------------------------------
__global__ __launch_bounds__(256) void fc_kernel(
    const unsigned short* __restrict__ y4, const float* __restrict__ fcw,
    const float* __restrict__ fcb, const float* __restrict__ cbuf,
    float* __restrict__ out) {
  int bb = blockIdx.x;
  int t = threadIdx.x;
  int cc = t & 31;                               // channel (coalesced y4 reads)
  int p0 = t >> 5;                               // pixel phase 0..7
  float acc[10] = {};
  for (int p = p0; p < 1024; p += 8) {
    int y = p >> 5, xx = p & 31;
    float a = bf2f(y4[((size_t)((bb * HP + y + 1) * HP + xx + 1)) * 32 + cc]);
    int wi = cc * 1024 + p;                      // NCHW flat index
#pragma unroll
    for (int j = 0; j < 10; ++j) acc[j] += a * fcw[j * 32768 + wi];
  }
#pragma unroll
  for (int j = 0; j < 10; ++j) {
    float s = acc[j];
#pragma unroll
    for (int msk = 16; msk; msk >>= 1) s += __shfl_xor(s, msk, 32);
    acc[j] = s;
  }
  __shared__ float red[8][10];
  int wave = t >> 5, lane = t & 31;
  if (lane == 0) {
#pragma unroll
    for (int j = 0; j < 10; ++j) red[wave][j] = acc[j];
  }
  __syncthreads();
  if (t < 10) {
    float s = 0.0f;
#pragma unroll
    for (int w = 0; w < 8; ++w) s += red[w][t];
    out[bb * 10 + t] = cbuf[bb] * s + fcb[t];
  }
}

// ------------------------------- launcher ----------------------------------
extern "C" void kernel_launch(void* const* d_in, const int* in_sizes, int n_in,
                              void* d_out, int out_size, void* d_ws, size_t ws_size,
                              hipStream_t stream) {
  const float* x    = (const float*)d_in[0];
  const float* w0   = (const float*)d_in[1];
  const float* w1   = (const float*)d_in[2];
  const float* w2   = (const float*)d_in[3];
  const float* w3   = (const float*)d_in[4];
  const float* fcw  = (const float*)d_in[5];
  const float* fcb  = (const float*)d_in[6];
  float* out = (float*)d_out;

  char* ws = (char*)d_ws;
  unsigned short* actA = (unsigned short*)ws;
  unsigned short* actB = (unsigned short*)(ws + ACT_BYTES);
  float* S    = (float*)(ws + 2 * ACT_BYTES);           // [4][1024]
  float* cbuf = S + 4 * 1024;
  unsigned short* wpack = (unsigned short*)((char*)(cbuf + 1024));

  // zero both activation buffers (halos) + S   (151,535,616 B = 9,470,976 uint4)
  zero_kernel<<<4096, 256, 0, stream>>>((uint4*)ws, 9470976u);
  pack_input_kernel<<<4096, 256, 0, stream>>>(x, actA);
  pack_weights_kernel<<<144, 256, 0, stream>>>(w0, w1, w2, w3, wpack);

  conv_kernel<<<8192, 256, 0, stream>>>(actA, actB, wpack + 0 * WPACK_PER_LAYER, S + 0 * 1024);
  conv_kernel<<<8192, 256, 0, stream>>>(actB, actA, wpack + 1 * WPACK_PER_LAYER, S + 1 * 1024);
  conv_kernel<<<8192, 256, 0, stream>>>(actA, actB, wpack + 2 * WPACK_PER_LAYER, S + 2 * 1024);
  conv_kernel<<<8192, 256, 0, stream>>>(actB, actA, wpack + 3 * WPACK_PER_LAYER, S + 3 * 1024);

  scale_kernel<<<4, 256, 0, stream>>>(S, cbuf);
  fc_kernel<<<1024, 256, 0, stream>>>(actA, fcw, fcb, cbuf, out);
}
